// SelfAttention_12103217840482
// MI455X (gfx1250) — matrix operand order repaired and checked by reading, not compiled, not run on hardware
//
#include <hip/hip_runtime.h>
#include <stdint.h>

typedef __attribute__((ext_vector_type(16))) _Float16 v16h;
typedef __attribute__((ext_vector_type(8)))  _Float16 v8h;
typedef __attribute__((ext_vector_type(4)))  _Float16 v4h;
typedef __attribute__((ext_vector_type(8)))  float    v8f;
typedef __attribute__((ext_vector_type(4)))  float    v4f;

#define BM 128
#define BN 256
#define BK 32
#define LDSP 40   // LDS row pitch in halves: 80 B (multiple of 16 B)

// Async DMA: global -> LDS, 16 B per lane, tracked by ASYNCcnt.
__device__ __forceinline__ void async_b128(uint32_t lds, const void* g) {
    asm volatile("global_load_async_to_lds_b128 %0, %1, off"
                 :: "v"(lds), "v"((uint64_t)(uintptr_t)g) : "memory");
}

// Unified f16 WMMA GEMM: C[M,N] = A[M,K] * B[N,K]^T where both operands are
// row-major along K.  Tile fills are pure 16B async DMA into double-buffered
// LDS; 8 waves in a 2x4 grid, each owning a 64x64 tile (16 WMMAs/K-step).
//  BIAS : add bias[n] (f32)
//  OUT16: store f16, else f32
//  OUTT : store transposed f16 as Vt[b][h][t] (hardcoded S=2048, D=1024)
template<bool BIAS, bool OUT16, bool OUTT>
__global__ __launch_bounds__(256)
void gemm_f16_wmma(const _Float16* __restrict__ Ap, const _Float16* __restrict__ Bp,
                   void* __restrict__ Cp, const float* __restrict__ bias,
                   int K, int lda, int ldb, int ldc,
                   long long sA, long long sB, long long sC)
{
    __shared__ _Float16 As[2][BM][LDSP];   // 20480 B
    __shared__ _Float16 Bs[2][BN][LDSP];   // 40960 B

    const int tid   = threadIdx.x;
    const int lane  = tid & 31;
    const int wid   = tid >> 5;        // 0..7
    const int wm    = wid >> 2;        // 0..1 -> 64-row half
    const int wn    = wid & 3;         // 0..3 -> 64-col quarter
    const int lhalf = lane >> 4;       // half-wave (WMMA layout)
    const int l16   = lane & 15;

    const int m0 = blockIdx.x * BM;
    const int n0 = blockIdx.y * BN;
    const int z  = blockIdx.z;

    const _Float16* A = Ap + (size_t)((long long)z * sA);
    const _Float16* B = Bp + (size_t)((long long)z * sB);

    // Tile chunking: A = 512 x 16B (2/thread), B = 1024 x 16B (4/thread).
    const int trow = tid >> 2;         // 0..63
    const int to   = (tid & 3) * 8;    // 0,8,16,24 halves

    auto issue = [&](int kt, int buf) {
        const size_t kq = (size_t)kt * BK + to;
        async_b128((uint32_t)(uintptr_t)&As[buf][trow][to],
                   A + (size_t)(m0 + trow) * lda + kq);
        async_b128((uint32_t)(uintptr_t)&As[buf][trow + 64][to],
                   A + (size_t)(m0 + trow + 64) * lda + kq);
        #pragma unroll
        for (int r = 0; r < 4; r++)
            async_b128((uint32_t)(uintptr_t)&Bs[buf][trow + r * 64][to],
                       B + (size_t)(n0 + trow + r * 64) * ldb + kq);
    };

    v8f acc[4][4] = {};
    const int nk = K / BK;

    auto compute = [&](int cur) {
        // A 16x32 f16: lane<16 holds K 0..7/16..23, lane>=16 holds K 8..15/24..31
        v16h afrag[4], bfrag[4];
        #pragma unroll
        for (int i = 0; i < 4; i++) {
            int row = wm * 64 + i * 16 + l16;
            v8h lo = *(const v8h*)&As[cur][row][lhalf * 8];
            v8h hi = *(const v8h*)&As[cur][row][16 + lhalf * 8];
            afrag[i] = __builtin_shufflevector(lo, hi, 0,1,2,3,4,5,6,7,8,9,10,11,12,13,14,15);
        }
        // B 32x16 f16: lane<16 holds K 0..7/16..23, lane>=16 holds K 8..15/24..31, col=lane%16
        #pragma unroll
        for (int j = 0; j < 4; j++) {
            int col = wn * 64 + j * 16 + l16;
            v8h lo = *(const v8h*)&Bs[cur][col][lhalf * 8];
            v8h hi = *(const v8h*)&Bs[cur][col][16 + lhalf * 8];
            bfrag[j] = __builtin_shufflevector(lo, hi, 0,1,2,3,4,5,6,7,8,9,10,11,12,13,14,15);
        }
        #pragma unroll
        for (int i = 0; i < 4; i++)
            #pragma unroll
            for (int j = 0; j < 4; j++)
                acc[i][j] = __builtin_amdgcn_wmma_f32_16x16x32_f16(
                    false, afrag[i], false, bfrag[j], (short)0, acc[i][j], false, false);
    };

    issue(0, 0);                                   // prologue: tile 0 -> buf 0

    // steady state: branch-free; prefetch kt+1 while computing kt
    for (int kt = 0; kt < nk - 1; ++kt) {
        const int cur = kt & 1;
        __syncthreads();                           // buf cur^1 free (iter kt-1 reads done)
        issue(kt + 1, cur ^ 1);
        asm volatile("s_wait_asynccnt 0x6" ::: "memory");  // tile kt landed (in-order)
        __syncthreads();                           // visible to all waves
        compute(cur);
    }
    // peeled last tile
    asm volatile("s_wait_asynccnt 0x0" ::: "memory");
    __syncthreads();
    compute((nk - 1) & 1);

    // -------- epilogue: C VGPR v -> row v + 8*lhalf in tile, col = lane%16 ----
    #pragma unroll
    for (int i = 0; i < 4; i++) {
        #pragma unroll
        for (int j = 0; j < 4; j++) {
            int col = n0 + wn * 64 + j * 16 + l16;
            float bv = BIAS ? bias[col] : 0.0f;
            if (OUTT) {
                // Vt[b][h=col][t]: 8 accumulator rows are t-contiguous -> 16B store
                int t0 = m0 + wm * 64 + i * 16 + 8 * lhalf;
                int b  = t0 >> 11, t = t0 & 2047;        // S = 2048
                v8h pk;
                #pragma unroll
                for (int v = 0; v < 8; v++) pk[v] = (_Float16)(acc[i][j][v] + bv);
                _Float16* C = (_Float16*)Cp;
                *(v8h*)&C[((size_t)b * 1024 + col) * 2048 + t] = pk;   // D = 1024
            } else {
                #pragma unroll
                for (int v = 0; v < 8; v++) {
                    int row = m0 + wm * 64 + i * 16 + v + 8 * lhalf;
                    float val = acc[i][j][v] + bv;
                    if (OUT16) {
                        _Float16* C = (_Float16*)Cp + (size_t)((long long)z * sC);
                        C[(size_t)row * ldc + col] = (_Float16)val;
                    } else {
                        float* C = (float*)Cp + (size_t)((long long)z * sC);
                        C[(size_t)row * ldc + col] = val;
                    }
                }
            }
        }
    }
}

// x (f32) -> xh (f16), flat, one float4 per thread.
__global__ __launch_bounds__(256)
void f32_to_f16(const float* __restrict__ in, _Float16* __restrict__ out)
{
    size_t i = (size_t)blockIdx.x * 256 + threadIdx.x;
    v4f v = ((const v4f*)in)[i];
    v4h h;
    h.x = (_Float16)v.x; h.y = (_Float16)v.y;
    h.z = (_Float16)v.z; h.w = (_Float16)v.w;
    ((v4h*)out)[i] = h;
}

// W[k][n] f32 -> Wt[n][k] f16, LDS-tiled 32x32 transpose.
__global__ __launch_bounds__(256)
void w_transpose_f16(const float* __restrict__ W, _Float16* __restrict__ Wt, int D)
{
    __shared__ float t[32][33];
    const int bx = blockIdx.x * 32;        // n tile
    const int by = blockIdx.y * 32;        // k tile
    const int tx = threadIdx.x & 31, ty = threadIdx.x >> 5;
    #pragma unroll
    for (int r = 0; r < 32; r += 8)
        t[ty + r][tx] = W[(size_t)(by + ty + r) * D + bx + tx];
    __syncthreads();
    #pragma unroll
    for (int r = 0; r < 32; r += 8)
        Wt[(size_t)(bx + ty + r) * D + by + tx] = (_Float16)t[tx][ty + r];
}

// Row softmax over 2048 f32 scores; rewrites the row IN PLACE as f16 probs at
// the same row base (stride 4096 halves == 2048 f32), so rows never alias.
__global__ __launch_bounds__(256)
void softmax_rows(float* __restrict__ sc)
{
    __shared__ float red[256];
    const int tid = threadIdx.x;
    const size_t row = blockIdx.x;
    const v4f* r4 = (const v4f*)(sc + row * 2048);
    v4h*       p4 = (v4h*)((_Float16*)sc + row * 4096);

    v4f a = r4[tid], b = r4[tid + 256];
    float m = fmaxf(fmaxf(fmaxf(a.x, a.y), fmaxf(a.z, a.w)),
                    fmaxf(fmaxf(b.x, b.y), fmaxf(b.z, b.w)));
    red[tid] = m; __syncthreads();
    for (int s = 128; s > 0; s >>= 1) {
        if (tid < s) red[tid] = fmaxf(red[tid], red[tid + s]);
        __syncthreads();
    }
    m = red[0]; __syncthreads();

    a.x = __expf(a.x - m); a.y = __expf(a.y - m);
    a.z = __expf(a.z - m); a.w = __expf(a.w - m);
    b.x = __expf(b.x - m); b.y = __expf(b.y - m);
    b.z = __expf(b.z - m); b.w = __expf(b.w - m);
    float s = (a.x + a.y) + (a.z + a.w) + (b.x + b.y) + (b.z + b.w);
    red[tid] = s; __syncthreads();
    for (int st = 128; st > 0; st >>= 1) {
        if (tid < st) red[tid] += red[tid + st];
        __syncthreads();
    }
    float inv = 1.0f / red[0];
    v4h ha, hb;
    ha.x = (_Float16)(a.x * inv); ha.y = (_Float16)(a.y * inv);
    ha.z = (_Float16)(a.z * inv); ha.w = (_Float16)(a.w * inv);
    hb.x = (_Float16)(b.x * inv); hb.y = (_Float16)(b.y * inv);
    hb.z = (_Float16)(b.z * inv); hb.w = (_Float16)(b.w * inv);
    p4[tid] = ha; p4[tid + 256] = hb;
}

extern "C" void kernel_launch(void* const* d_in, const int* in_sizes, int n_in,
                              void* d_out, int out_size, void* d_ws, size_t ws_size,
                              hipStream_t stream)
{
    (void)in_sizes; (void)n_in; (void)out_size; (void)ws_size;
    const float* x  = (const float*)d_in[0];
    const float* wq = (const float*)d_in[1];
    const float* bq = (const float*)d_in[2];
    const float* wk = (const float*)d_in[3];
    const float* bk = (const float*)d_in[4];
    const float* wv = (const float*)d_in[5];
    const float* bv = (const float*)d_in[6];

    const int Bb = 4, S = 2048, D = 1024;
    const long long MS = (long long)Bb * S;       // 8192

    // Workspace: xh(16M) | Wt q/k/v (2M each) | Qh(16M) | Kh(16M) | Vt(16M) | Sc(64M)
    _Float16* xh  = (_Float16*)d_ws;
    _Float16* Wtq = xh  + (size_t)MS * D;
    _Float16* Wtk = Wtq + (size_t)D * D;
    _Float16* Wtv = Wtk + (size_t)D * D;
    _Float16* Qh  = Wtv + (size_t)D * D;
    _Float16* Kh  = Qh  + (size_t)MS * D;
    _Float16* Vt  = Kh  + (size_t)MS * D;         // layout [b][h][t]
    float*    Sc  = (float*)(Vt + (size_t)MS * D);

    dim3 blk(256);

    // 0) one-time precision / layout conversion
    f32_to_f16<<<dim3((unsigned)(MS * D / (4 * 256))), blk, 0, stream>>>(x, xh);
    dim3 gt(D / 32, D / 32);
    w_transpose_f16<<<gt, blk, 0, stream>>>(wq, Wtq, D);
    w_transpose_f16<<<gt, blk, 0, stream>>>(wk, Wtk, D);
    w_transpose_f16<<<gt, blk, 0, stream>>>(wv, Wtv, D);

    // 1) projections: Q,K row-major f16; V written pre-transposed as Vt[b][h][t]
    dim3 g1((unsigned)(MS / BM), D / BN, 1);
    gemm_f16_wmma<true, true, false><<<g1, blk, 0, stream>>>(
        xh, Wtq, Qh, bq, D, D, D, D, 0LL, 0LL, 0LL);
    gemm_f16_wmma<true, true, false><<<g1, blk, 0, stream>>>(
        xh, Wtk, Kh, bk, D, D, D, D, 0LL, 0LL, 0LL);
    gemm_f16_wmma<true, true, true><<<g1, blk, 0, stream>>>(
        xh, Wtv, Vt, bv, D, D, D, D, 0LL, 0LL, 0LL);

    // 2) scores[b] = Q[b] @ K[b]^T   (K rows are [t][h] == [n][k])
    dim3 g2(S / BM, S / BN, Bb);
    gemm_f16_wmma<false, false, false><<<g2, blk, 0, stream>>>(
        Qh, Kh, Sc, nullptr, D, D, D, S,
        (long long)S * D, (long long)S * D, (long long)S * S);

    // 3) row softmax (f32 -> f16 probs, in place)
    softmax_rows<<<dim3((unsigned)MS), blk, 0, stream>>>(Sc);

    // 4) out[b] = P[b] @ V[b]   (P rows stride 4096 halves; Vt rows are [h][t])
    dim3 g4(S / BM, D / BN, Bb);
    gemm_f16_wmma<false, false, false><<<g4, blk, 0, stream>>>(
        (const _Float16*)Sc, Vt, (float*)d_out, nullptr, S,
        2 * S, S, D,
        (long long)S * (2 * S), (long long)D * S, (long long)S * D);
}
